// InsclaeNonLocalAttention_52647709114616
// MI455X (gfx1250) — matrix operand order repaired and checked by reading, not compiled, not run on hardware
//
#include <hip/hip_runtime.h>

typedef __attribute__((ext_vector_type(16))) _Float16 v16h;
typedef __attribute__((ext_vector_type(8)))  float    v8f;
typedef __attribute__((ext_vector_type(4)))  unsigned int u32x4;

union AFrag { v16h v; _Float16 e[16]; u32x4 q[2]; unsigned int d[8]; };
union CFrag { v8f  v; float e[8]; };

#define Bb 4
#define Nn 9216
#define Cc 64
#define ICc 32
#define QSCALE 8.656170245333781f   /* 6 * log2(e): softmax factor folded into Q */

// ---------------------------------------------------------------------------
// helpers
// ---------------------------------------------------------------------------
__device__ __forceinline__ float fast_exp2(float x) {        // v_exp_f32
    return __builtin_amdgcn_exp2f(x);
}
__device__ __forceinline__ float swz16f(float x) {           // xor-16 lane swap
    return __int_as_float(__builtin_amdgcn_ds_swizzle(__float_as_int(x), 0x401F));
}
__device__ __forceinline__ unsigned swz16u(unsigned x) {
    return (unsigned)__builtin_amdgcn_ds_swizzle((int)x, 0x401F);
}
__device__ __forceinline__ unsigned pack_h2(float a, float b) {
    union { _Float16 h[2]; unsigned u; } t;
    t.h[0] = (_Float16)a; t.h[1] = (_Float16)b;
    return t.u;
}
// async DMA: 16B per lane, global -> LDS (CDNA5 GLOBAL_LOAD_ASYNC_TO_LDS_B128)
__device__ __forceinline__ void async_copy_b128(unsigned lds_off,
                                                unsigned long long gaddr) {
    asm volatile("global_load_async_to_lds_b128 %0, %1, off"
                 :: "v"(lds_off), "v"(gaddr) : "memory");
}
__device__ __forceinline__ void wait_async0() {
    asm volatile("s_wait_asynccnt 0x0" ::: "memory");
}

// ---------------------------------------------------------------------------
// Phase 1: theta/phi/g projections.  Qh = (x@w_theta + b)*QSCALE (f16),
// Kh = x@w_phi + b (f16), Vtg = (x@w_g + b) stored TRANSPOSED [b][dim][N].
// ---------------------------------------------------------------------------
__global__ __launch_bounds__(96) void proj_qkv(
    const float* __restrict__ x,
    const float* __restrict__ w_theta, const float* __restrict__ b_theta,
    const float* __restrict__ w_phi,   const float* __restrict__ b_phi,
    const float* __restrict__ w_g,     const float* __restrict__ b_g,
    _Float16* __restrict__ Qh, _Float16* __restrict__ Kh,
    _Float16* __restrict__ Vtg)
{
    __shared__ float xs[64];
    const int row = blockIdx.x;            // 0 .. B*N-1
    const int tid = threadIdx.x;
    if (tid < 64) xs[tid] = x[(size_t)row * Cc + tid];
    __syncthreads();
    const int mat = tid >> 5;              // uniform per wave
    const int col = tid & 31;
    const float* W  = (mat == 0) ? w_theta : (mat == 1) ? w_phi : w_g;
    const float* Bv = (mat == 0) ? b_theta : (mat == 1) ? b_phi : b_g;
    float s = Bv[col];
    #pragma unroll
    for (int c = 0; c < Cc; ++c) s += xs[c] * W[c * ICc + col];
    if (mat == 0) {
        Qh[(size_t)row * ICc + col] = (_Float16)(s * QSCALE);
    } else if (mat == 1) {
        Kh[(size_t)row * ICc + col] = (_Float16)s;
    } else {
        const int bi = row / Nn, n = row % Nn;
        Vtg[((size_t)bi * ICc + col) * Nn + n] = (_Float16)s;
    }
}

// ---------------------------------------------------------------------------
// Phase 2: flash attention, transposed formulation.
// 1 WG = (batch, 128 queries), 8 waves x 16 queries. 32-key tiles, double-
// buffered LDS filled by async b128 copies; softmax reductions are in-register
// (keys live on the M axis of S' = K_tile * Q^T).
// ---------------------------------------------------------------------------
__global__ __launch_bounds__(256) void flash_attn(
    const _Float16* __restrict__ Qh, const _Float16* __restrict__ Kh,
    const _Float16* __restrict__ Vtg, float* __restrict__ Of)
{
    __shared__ _Float16 Ks[2][32 * 32];   // [key][dim]
    __shared__ _Float16 Vs[2][32 * 32];   // [dim][key]  (pre-transposed)

    const int b    = blockIdx.x / (Nn / 128);
    const int qblk = blockIdx.x % (Nn / 128);
    const int tid  = threadIdx.x;
    const int wave = tid >> 5;
    const int lane = tid & 31;
    const int lm   = lane & 15;
    const int half = lane >> 4;
    const size_t bbase = (size_t)b * Nn;

    // B-fragment of Q^T: lane holds query column n=lm, dims {8h..8h+7, 16+8h..16+8h+7}
    const int q0 = qblk * 128 + wave * 16;
    AFrag bQ;
    {
        const _Float16* qp = Qh + (bbase + q0 + lm) * ICc + 8 * half;
        bQ.q[0] = *(const u32x4*)(qp);
        bQ.q[1] = *(const u32x4*)(qp + 16);
    }

    CFrag O0, O1;
    #pragma unroll
    for (int r = 0; r < 8; ++r) { O0.e[r] = 0.f; O1.e[r] = 0.f; }
    float mrun = -1e30f, lrun = 0.f;

    // cooperative async fill: 256 threads x 16B = K tile (2KB) + Vt tile (2KB)
    const int  ct   = tid & 127;
    const int  crow = ct >> 2;            // key (K) or dim (Vt), 0..31
    const int  coff = (ct & 3) * 8;       // half-offset within 32-half row
    unsigned lds0, lds1;
    unsigned long long gsrc, gstep;
    if (tid < 128) {
        lds0 = (unsigned)(size_t)&Ks[0][crow * 32 + coff];
        lds1 = (unsigned)(size_t)&Ks[1][crow * 32 + coff];
        gsrc = (unsigned long long)(size_t)(Kh + (bbase + crow) * ICc + coff);
        gstep = 32ull * ICc * sizeof(_Float16);          // next 32-key tile
    } else {
        lds0 = (unsigned)(size_t)&Vs[0][crow * 32 + coff];
        lds1 = (unsigned)(size_t)&Vs[1][crow * 32 + coff];
        gsrc = (unsigned long long)(size_t)(Vtg + ((size_t)b * ICc + crow) * Nn + coff);
        gstep = 32ull * sizeof(_Float16);                // next 32 keys in row
    }

    async_copy_b128(lds0, gsrc);                          // prologue: tile 0
    unsigned long long gnext = gsrc + gstep;

    const int NT = Nn / 32;
    for (int it = 0; it < NT; ++it) {
        wait_async0();          // my chunk of buf[it&1] landed in LDS
        __syncthreads();        // everyone's chunks landed; prev reads done
        if (it + 1 < NT) {      // overlap next tile's DMA with compute
            async_copy_b128((it & 1) ? lds0 : lds1, gnext);
            gnext += gstep;
        }
        const _Float16* K0 = Ks[it & 1];
        const _Float16* V0 = Vs[it & 1];

        // A-fragments (ka: interleaved K-chunks {8h..8h+7, 16+8h..16+8h+7}; va: contiguous 16h..16h+15)
        AFrag ka0, ka1, va0, va1;
        {
            const _Float16* kp = K0 + lm * 32 + 8 * half;
            ka0.q[0] = *(const u32x4*)(kp);
            ka0.q[1] = *(const u32x4*)(kp + 16);
            const _Float16* kq = K0 + (lm + 16) * 32 + 8 * half;
            ka1.q[0] = *(const u32x4*)(kq);
            ka1.q[1] = *(const u32x4*)(kq + 16);
            const _Float16* vp = V0 + lm * 32 + 16 * half;
            va0.q[0] = *(const u32x4*)(vp);
            va0.q[1] = *(const u32x4*)(vp + 8);
            const _Float16* vq = V0 + (lm + 16) * 32 + 16 * half;
            va1.q[0] = *(const u32x4*)(vq);
            va1.q[1] = *(const u32x4*)(vq + 8);
        }

        // S' = K_tile(32x32) * Q^T(32x16): keys on M, queries on N (lanes)
        CFrag S0, S1;
        #pragma unroll
        for (int r = 0; r < 8; ++r) { S0.e[r] = 0.f; S1.e[r] = 0.f; }
        S0.v = __builtin_amdgcn_wmma_f32_16x16x32_f16(
                   false, ka0.v, false, bQ.v, (short)0, S0.v, false, false);
        S1.v = __builtin_amdgcn_wmma_f32_16x16x32_f16(
                   false, ka1.v, false, bQ.v, (short)0, S1.v, false, false);

        // online softmax per query (per lane): in-register tree + 1 lane swap
        float mx = fmaxf(S0.e[0], S1.e[0]);
        #pragma unroll
        for (int r = 1; r < 8; ++r) mx = fmaxf(mx, fmaxf(S0.e[r], S1.e[r]));
        mx = fmaxf(mx, swz16f(mx));
        const float mnew = fmaxf(mrun, mx);
        const float a    = fast_exp2(mrun - mnew);
        float rs = 0.f;
        #pragma unroll
        for (int r = 0; r < 8; ++r) {
            float p0 = fast_exp2(S0.e[r] - mnew);
            float p1 = fast_exp2(S1.e[r] - mnew);
            S0.e[r] = p0; S1.e[r] = p1;
            rs += p0 + p1;
        }
        rs += swz16f(rs);
        lrun = lrun * a + rs;
        mrun = mnew;
        #pragma unroll
        for (int r = 0; r < 8; ++r) { O0.e[r] *= a; O1.e[r] *= a; }

        // Build B-fragment of P^T via packed half-wave exchange (8 swizzles)
        AFrag bP;
        #pragma unroll
        for (int j = 0; j < 4; ++j) {
            unsigned pk0 = pack_h2(S0.e[2 * j], S0.e[2 * j + 1]); // keys 8h+2j..
            unsigned pk1 = pack_h2(S1.e[2 * j], S1.e[2 * j + 1]); // keys 16+8h+2j..
            unsigned x0 = swz16u(pk0);
            unsigned x1 = swz16u(pk1);
            bP.d[j]     = half ? x1 : pk0;   // keys 16h+0..7
            bP.d[4 + j] = half ? pk1 : x0;   // keys 16h+8..15
        }

        // O'^T += V^T(32dims x 32keys) * P^T(32keys x 16queries)
        O0.v = __builtin_amdgcn_wmma_f32_16x16x32_f16(
                   false, va0.v, false, bP.v, (short)0, O0.v, false, false);
        O1.v = __builtin_amdgcn_wmma_f32_16x16x32_f16(
                   false, va1.v, false, bP.v, (short)0, O1.v, false, false);
    }

    // normalize, store O: lane owns query q0+lm; frag rows are dims
    const float inv = 1.0f / lrun;
    float* op = Of + (bbase + q0 + lm) * ICc;
    #pragma unroll
    for (int r = 0; r < 8; ++r) {
        op[r + 8 * half]      = O0.e[r] * inv;
        op[16 + r + 8 * half] = O1.e[r] * inv;
    }
}

// ---------------------------------------------------------------------------
// Phase 3: out = O @ w_y + b_y   ([B*N,32] x [32,64])
// ---------------------------------------------------------------------------
__global__ __launch_bounds__(64) void proj_out(
    const float* __restrict__ Of, const float* __restrict__ w_y,
    const float* __restrict__ b_y, float* __restrict__ out)
{
    __shared__ float os[32];
    const int row = blockIdx.x;
    const int col = threadIdx.x;
    if (col < 32) os[col] = Of[(size_t)row * ICc + col];
    __syncthreads();
    float s = b_y[col];
    #pragma unroll
    for (int ic = 0; ic < ICc; ++ic) s += os[ic] * w_y[ic * Cc + col];
    out[(size_t)row * Cc + col] = s;
}

// ---------------------------------------------------------------------------
extern "C" void kernel_launch(void* const* d_in, const int* in_sizes, int n_in,
                              void* d_out, int out_size, void* d_ws, size_t ws_size,
                              hipStream_t stream) {
    const float* x       = (const float*)d_in[0];
    const float* w_theta = (const float*)d_in[1];
    const float* b_theta = (const float*)d_in[2];
    const float* w_phi   = (const float*)d_in[3];
    const float* b_phi   = (const float*)d_in[4];
    const float* w_g     = (const float*)d_in[5];
    const float* b_g     = (const float*)d_in[6];
    const float* w_y     = (const float*)d_in[7];
    const float* b_y     = (const float*)d_in[8];
    float* out = (float*)d_out;

    const size_t qkv_bytes = (size_t)Bb * Nn * ICc * sizeof(_Float16); // 2.25 MB
    char* ws = (char*)d_ws;
    _Float16* Qh  = (_Float16*)(ws);
    _Float16* Kh  = (_Float16*)(ws + qkv_bytes);
    _Float16* Vtg = (_Float16*)(ws + 2 * qkv_bytes);
    float*    Of  = (float*)   (ws + 3 * qkv_bytes);

    proj_qkv<<<Bb * Nn, 96, 0, stream>>>(x, w_theta, b_theta, w_phi, b_phi,
                                         w_g, b_g, Qh, Kh, Vtg);
    flash_attn<<<Bb * (Nn / 128), 256, 0, stream>>>(Qh, Kh, Vtg, Of);
    proj_out<<<Bb * Nn, 64, 0, stream>>>(Of, w_y, b_y, out);
}
